// SemanticAnchorAttention_73761768341557
// MI455X (gfx1250) — matrix operand order repaired and checked by reading, not compiled, not run on hardware
//
#include <hip/hip_runtime.h>
#include <hip/hip_bf16.h>

typedef __attribute__((ext_vector_type(16))) __bf16 v16bf;
typedef __attribute__((ext_vector_type(8)))  __bf16 v8bf;
typedef __attribute__((ext_vector_type(8)))  float  v8f;

#define DIMC 384
#define NHEAD 8
#define HDIM 48
#define MANCH 16
#define BATCH 16
#define HGT 64
#define WID 64
#define NTOK 4096
#define ROWS (BATCH * NTOK)   // 65536
#define SCALE 0.14433756729740643f  // 1/sqrt(48)

// ---------------------------------------------------------------------------
// fp32 -> bf16 with transpose: in [K, cols] row-major -> out [cols, K]
// (weights stored K-major so B fragments load contiguously)
// ---------------------------------------------------------------------------
__global__ void cvt_transpose_kernel(const float* __restrict__ in,
                                     __bf16* __restrict__ out,
                                     int rowsK, int cols) {
    int i = blockIdx.x * blockDim.x + threadIdx.x;
    if (i >= rowsK * cols) return;
    int u = i / cols, c = i - u * cols;
    out[(long)c * rowsK + u] = (__bf16)in[i];
}

// ---------------------------------------------------------------------------
// Saliency gate: sal = sigmoid(x . w_s + b); emit bf16 x and bf16 x*sal.
// One wave per token; 12 channels per lane; shfl_xor reduction.
// ---------------------------------------------------------------------------
__global__ void saliency_kernel(const float* __restrict__ x,
                                const float* __restrict__ sw,
                                const float* __restrict__ sb,
                                __bf16* __restrict__ x_bf,
                                __bf16* __restrict__ xsal_bf) {
    int wave = threadIdx.x >> 5;
    int lane = threadIdx.x & 31;
    long t = (long)blockIdx.x * 8 + wave;
    const float* xr = x + t * DIMC;
    float xv[12];
    float part = 0.f;
    #pragma unroll
    for (int j = 0; j < 12; ++j) {
        int c = lane + 32 * j;
        xv[j] = xr[c];
        part += xv[j] * sw[c];
    }
    #pragma unroll
    for (int off = 16; off; off >>= 1) part += __shfl_xor(part, off);
    float sal = 1.f / (1.f + __expf(-(part + sb[0])));
    #pragma unroll
    for (int j = 0; j < 12; ++j) {
        int c = lane + 32 * j;
        x_bf[t * DIMC + c]    = (__bf16)xv[j];
        xsal_bf[t * DIMC + c] = (__bf16)(xv[j] * sal);
    }
}

// ---------------------------------------------------------------------------
// bf16 WMMA GEMM with transposed weights:
//   C[rows, COLS] = A[rows, K] @ W  where WT = W^T stored [COLS, K].
// Block = 256 threads = 8 waves. Wave (r,c) with r=waveid&3, c=waveid>>2
// computes a 16x32 strip: rows [blockIdx.x*64 + r*16), cols [blockIdx.y*64
// + c*32). Two B fragments / two accumulators per wave -> interleaved WMMAs
// (no back-to-back RAW hazard) and A fragment reused twice.
//
// Fragment layouts used here (bf16, 16x16x32; A and B share contiguous k order):
//   A: lane group g=lane/16, row=lane%16;
//      elems j=0..15 -> K = k0 + g*16 + j
//   B: col = lane%16; elems j=0..15 -> K = k0 + g*16 + j
//      With WT layout those 16 elems are contiguous: one 32B (2x b128) load.
//   C/D: elem i -> row = row0 + g*8 + i, col = col0 + lane%16
// ---------------------------------------------------------------------------
template<int K, int COLS, int OUT_BF16>
__global__ void wmma_gemm_bf16_kernel(const __bf16* __restrict__ A,
                                      const __bf16* __restrict__ WT,
                                      void* __restrict__ Cout) {
    int waveid = threadIdx.x >> 5;
    int lane   = threadIdx.x & 31;
    int g      = lane >> 4;
    int lr     = lane & 15;
    long row0 = (long)blockIdx.x * 64 + (waveid & 3) * 16;
    int  col0 = blockIdx.y * 64 + (waveid >> 2) * 32;

    const __bf16* ap  = A  + (row0 + lr) * K + g * 16;
    const __bf16* bp0 = WT + (long)(col0 + lr) * K + g * 16;
    const __bf16* bp1 = bp0 + (long)16 * K;

    v8f acc0 = {}, acc1 = {};
    #pragma unroll
    for (int k0 = 0; k0 < K; k0 += 32) {
        v8bf alo = *(const v8bf*)(ap + k0);        // K = k0+g*16 .. +7
        v8bf ahi = *(const v8bf*)(ap + k0 + 8);    // K = k0+g*16+8 .. +15
        v16bf afrag = __builtin_shufflevector(alo, ahi,
            0, 1, 2, 3, 4, 5, 6, 7, 8, 9, 10, 11, 12, 13, 14, 15);

        v16bf bf0 = *(const v16bf*)(bp0 + k0);     // K = k0+g*16 .. +15 (32B)
        v16bf bf1 = *(const v16bf*)(bp1 + k0);

        acc0 = __builtin_amdgcn_wmma_f32_16x16x32_bf16(
            false, afrag, false, bf0, (short)0, acc0, false, false);
        acc1 = __builtin_amdgcn_wmma_f32_16x16x32_bf16(
            false, afrag, false, bf1, (short)0, acc1, false, false);
    }

    long obase = (row0 + g * 8) * (long)COLS + col0 + lr;
    if (OUT_BF16) {
        __bf16* C = (__bf16*)Cout;
        #pragma unroll
        for (int i = 0; i < 8; ++i) {
            C[obase + (long)i * COLS]      = (__bf16)acc0[i];
            C[obase + (long)i * COLS + 16] = (__bf16)acc1[i];
        }
    } else {
        float* C = (float*)Cout;
        #pragma unroll
        for (int i = 0; i < 8; ++i) {
            C[obase + (long)i * COLS]      = acc0[i];
            C[obase + (long)i * COLS + 16] = acc1[i];
        }
    }
}

// ---------------------------------------------------------------------------
// q0 = anchor_seeds @ w_q_agg  (batch-independent)
// ---------------------------------------------------------------------------
__global__ void qagg_kernel(const float* __restrict__ seeds,
                            const float* __restrict__ wq,
                            float* __restrict__ q0) {
    int i = blockIdx.x * blockDim.x + threadIdx.x;
    if (i >= MANCH * DIMC) return;
    int m = i / DIMC, c = i - m * DIMC;
    float s = 0.f;
    for (int u = 0; u < DIMC; ++u) s += seeds[m * DIMC + u] * wq[u * DIMC + c];
    q0[i] = s;
}

// ---------------------------------------------------------------------------
// Aggregation cross-attention: anchors1 = seeds + MHSA(seeds, x_sal).
// One block per (batch, head); each wave owns 2 queries; online softmax over
// 4096 keys with lane-striding + cross-lane merge.
// ---------------------------------------------------------------------------
__global__ void agg_attn_kernel(const float* __restrict__ q0,
                                const __bf16* __restrict__ kv,
                                const float* __restrict__ seeds,
                                float* __restrict__ anchors1) {
    int b = blockIdx.x >> 3;
    int h = blockIdx.x & 7;
    int wave = threadIdx.x >> 5;
    int lane = threadIdx.x & 31;

    for (int mq = 0; mq < 2; ++mq) {
        int m = wave * 2 + mq;
        float qv[HDIM];
        #pragma unroll
        for (int d = 0; d < HDIM; ++d) qv[d] = q0[m * DIMC + h * HDIM + d];

        float mmax = -3.0e38f, ssum = 0.f;
        float accv[HDIM];
        #pragma unroll
        for (int d = 0; d < HDIM; ++d) accv[d] = 0.f;

        for (int n = lane; n < NTOK; n += 32) {
            const __bf16* kp = kv + ((long)(b * NTOK + n)) * (2 * DIMC) + h * HDIM;
            const __bf16* vp = kp + DIMC;
            float l = 0.f;
            #pragma unroll
            for (int d = 0; d < HDIM; ++d) l += qv[d] * (float)kp[d];
            l *= SCALE;
            float nm = fmaxf(mmax, l);
            float c1 = __expf(mmax - nm);
            float p  = __expf(l - nm);
            ssum = ssum * c1 + p;
            #pragma unroll
            for (int d = 0; d < HDIM; ++d) accv[d] = accv[d] * c1 + p * (float)vp[d];
            mmax = nm;
        }
        for (int off = 16; off; off >>= 1) {
            float m2 = __shfl_xor(mmax, off);
            float s2 = __shfl_xor(ssum, off);
            float nm = fmaxf(mmax, m2);
            float c1 = __expf(mmax - nm), c2 = __expf(m2 - nm);
            ssum = ssum * c1 + s2 * c2;
            #pragma unroll
            for (int d = 0; d < HDIM; ++d) {
                float a2 = __shfl_xor(accv[d], off);
                accv[d] = accv[d] * c1 + a2 * c2;
            }
            mmax = nm;
        }
        float inv = 1.f / ssum;
        long ob = ((long)(b * MANCH + m)) * DIMC + h * HDIM;
        anchors1[ob + lane] = seeds[m * DIMC + h * HDIM + lane] + accv[lane] * inv;
        if (lane < HDIM - 32)
            anchors1[ob + 32 + lane] =
                seeds[m * DIMC + h * HDIM + 32 + lane] + accv[32 + lane] * inv;
    }
}

// ---------------------------------------------------------------------------
// Anchor self-attention + kv_bcast projection, one block per batch.
// ---------------------------------------------------------------------------
__global__ void self_attn_kernel(const float* __restrict__ a1g,
                                 const float* __restrict__ wq,
                                 const float* __restrict__ wkv,
                                 const float* __restrict__ wkvb,
                                 float* __restrict__ kvb_out) {
    __shared__ float ks[MANCH * DIMC];
    __shared__ float vs[MANCH * DIMC];
    int b = blockIdx.x;
    int t = threadIdx.x;
    const float* a1 = a1g + (long)b * MANCH * DIMC;

    for (int i = t; i < MANCH * 2 * DIMC; i += 256) {
        int m = i / (2 * DIMC), cc = i - m * 2 * DIMC;
        float s = 0.f;
        for (int u = 0; u < DIMC; ++u) s += a1[m * DIMC + u] * wkv[u * 2 * DIMC + cc];
        if (cc < DIMC) ks[m * DIMC + cc] = s;
        else           vs[m * DIMC + cc - DIMC] = s;
    }
    __syncthreads();

    float o[HDIM];
    int h = t >> 4, m = t & 15;
    if (t < NHEAD * MANCH) {
        float qreg[HDIM];
        #pragma unroll
        for (int d = 0; d < HDIM; ++d) {
            float s = 0.f;
            for (int u = 0; u < DIMC; ++u) s += a1[m * DIMC + u] * wq[u * DIMC + h * HDIM + d];
            qreg[d] = s;
        }
        float lg[MANCH];
        float mx = -3.0e38f;
        #pragma unroll
        for (int j = 0; j < MANCH; ++j) {
            float s = 0.f;
            #pragma unroll
            for (int d = 0; d < HDIM; ++d) s += qreg[d] * ks[j * DIMC + h * HDIM + d];
            lg[j] = s * SCALE;
            mx = fmaxf(mx, lg[j]);
        }
        float den = 0.f;
        #pragma unroll
        for (int j = 0; j < MANCH; ++j) { lg[j] = __expf(lg[j] - mx); den += lg[j]; }
        float inv = 1.f / den;
        #pragma unroll
        for (int d = 0; d < HDIM; ++d) {
            float s = 0.f;
            #pragma unroll
            for (int j = 0; j < MANCH; ++j) s += lg[j] * vs[j * DIMC + h * HDIM + d];
            o[d] = a1[m * DIMC + h * HDIM + d] + s * inv;
        }
    }
    __syncthreads();
    if (t < NHEAD * MANCH) {
        #pragma unroll
        for (int d = 0; d < HDIM; ++d) ks[m * DIMC + h * HDIM + d] = o[d]; // ks := anchors2
    }
    __syncthreads();

    for (int i = t; i < MANCH * 2 * DIMC; i += 256) {
        int m2 = i / (2 * DIMC), cc = i - m2 * 2 * DIMC;
        float s = 0.f;
        for (int u = 0; u < DIMC; ++u) s += ks[m2 * DIMC + u] * wkvb[u * 2 * DIMC + cc];
        kvb_out[(long)b * MANCH * 2 * DIMC + i] = s;
    }
}

// ---------------------------------------------------------------------------
// Broadcast attention (16 anchors) + depthwise 3x3 conv + combine -> bf16.
// ---------------------------------------------------------------------------
__global__ void bcast_combine_kernel(const float* __restrict__ x,
                                     const float* __restrict__ qb,
                                     const float* __restrict__ kvb,
                                     const float* __restrict__ dwk,
                                     __bf16* __restrict__ gl) {
    long tid = (long)blockIdx.x * blockDim.x + threadIdx.x;
    int h = (int)(tid & 7);
    long bn = tid >> 3;
    int n = (int)(bn & (NTOK - 1));
    int b = (int)(bn >> 12);

    float q[HDIM];
    const float* qp = qb + bn * DIMC + h * HDIM;
    #pragma unroll
    for (int d = 0; d < HDIM; ++d) q[d] = qp[d];

    float lg[MANCH];
    float mx = -3.0e38f;
    #pragma unroll
    for (int j = 0; j < MANCH; ++j) {
        const float* kp = kvb + ((long)b * MANCH + j) * (2 * DIMC) + h * HDIM;
        float s = 0.f;
        #pragma unroll
        for (int d = 0; d < HDIM; ++d) s += q[d] * kp[d];
        lg[j] = s * SCALE;
        mx = fmaxf(mx, lg[j]);
    }
    float den = 0.f;
    #pragma unroll
    for (int j = 0; j < MANCH; ++j) { lg[j] = __expf(lg[j] - mx); den += lg[j]; }
    float inv = 1.f / den;

    float outv[HDIM];
    #pragma unroll
    for (int d = 0; d < HDIM; ++d) outv[d] = 0.f;
    #pragma unroll
    for (int j = 0; j < MANCH; ++j) {
        const float* vp = kvb + ((long)b * MANCH + j) * (2 * DIMC) + DIMC + h * HDIM;
        float p = lg[j] * inv;
        #pragma unroll
        for (int d = 0; d < HDIM; ++d) outv[d] += p * vp[d];
    }

    int y = n >> 6, xp = n & 63;
    for (int d = 0; d < HDIM; ++d) {
        int c = h * HDIM + d;
        float cs = 0.f;
        #pragma unroll
        for (int dy = -1; dy <= 1; ++dy) {
            int yy = y + dy;
            if (yy < 0 || yy >= HGT) continue;
            #pragma unroll
            for (int dx = -1; dx <= 1; ++dx) {
                int xx = xp + dx;
                if (xx < 0 || xx >= WID) continue;
                cs += x[((long)b * NTOK + yy * WID + xx) * DIMC + c] *
                      dwk[c * 9 + (dy + 1) * 3 + (dx + 1)];
            }
        }
        gl[bn * DIMC + c] = (__bf16)(outv[d] + cs);
    }
}

// ---------------------------------------------------------------------------
// Host launcher
// ---------------------------------------------------------------------------
extern "C" void kernel_launch(void* const* d_in, const int* in_sizes, int n_in,
                              void* d_out, int out_size, void* d_ws, size_t ws_size,
                              hipStream_t stream) {
    const float* x          = (const float*)d_in[0];
    const float* saliency_w = (const float*)d_in[3];
    const float* saliency_b = (const float*)d_in[4];
    const float* seeds      = (const float*)d_in[5];
    const float* w_q_agg    = (const float*)d_in[6];
    const float* w_kv_agg   = (const float*)d_in[7];
    const float* w_q_self   = (const float*)d_in[8];
    const float* w_kv_self  = (const float*)d_in[9];
    const float* w_q_bcast  = (const float*)d_in[10];
    const float* w_kv_bcast = (const float*)d_in[11];
    const float* dw_kernel  = (const float*)d_in[12];
    const float* w_proj     = (const float*)d_in[13];
    float* out = (float*)d_out;

    char* base = (char*)d_ws;
    size_t off = 0;
    auto take = [&](size_t bytes) { char* p = base + off; off += (bytes + 255) & ~(size_t)255; return p; };

    __bf16* x_bf      = (__bf16*)take((size_t)ROWS * DIMC * 2);
    __bf16* xsal_bf   = (__bf16*)take((size_t)ROWS * DIMC * 2);
    __bf16* kvagg_bf  = (__bf16*)take((size_t)ROWS * 2 * DIMC * 2);
    float*  qb_f32    = (float*)take((size_t)ROWS * DIMC * 4);
    __bf16* gl_bf     = (__bf16*)take((size_t)ROWS * DIMC * 2);
    float*  q0_f32    = (float*)take((size_t)MANCH * DIMC * 4);
    float*  anchors1  = (float*)take((size_t)BATCH * MANCH * DIMC * 4);
    float*  kvb_f32   = (float*)take((size_t)BATCH * MANCH * 2 * DIMC * 4);
    __bf16* wkvaggT   = (__bf16*)take((size_t)DIMC * 2 * DIMC * 2);   // [768, 384]
    __bf16* wqbT      = (__bf16*)take((size_t)DIMC * DIMC * 2);       // [384, 384]
    __bf16* wprojT    = (__bf16*)take((size_t)DIMC * DIMC * 2);       // [384, 384]

    // 1) weight conversions to bf16, transposed to [cols, K]
    cvt_transpose_kernel<<<(DIMC * 2 * DIMC + 255) / 256, 256, 0, stream>>>(w_kv_agg, wkvaggT, DIMC, 2 * DIMC);
    cvt_transpose_kernel<<<(DIMC * DIMC + 255) / 256, 256, 0, stream>>>(w_q_bcast, wqbT, DIMC, DIMC);
    cvt_transpose_kernel<<<(DIMC * DIMC + 255) / 256, 256, 0, stream>>>(w_proj, wprojT, DIMC, DIMC);

    // 2) saliency gate + bf16 conversion of x and x*sal
    saliency_kernel<<<ROWS / 8, 256, 0, stream>>>(x, saliency_w, saliency_b, x_bf, xsal_bf);

    // 3) kv_agg = x_sal @ w_kv_agg  (WMMA, bf16 out)
    {
        dim3 grid(ROWS / 64, (2 * DIMC) / 64);
        wmma_gemm_bf16_kernel<DIMC, 2 * DIMC, 1><<<grid, 256, 0, stream>>>(xsal_bf, wkvaggT, kvagg_bf);
    }

    // 4) q0 = seeds @ w_q_agg
    qagg_kernel<<<(MANCH * DIMC + 255) / 256, 256, 0, stream>>>(seeds, w_q_agg, q0_f32);

    // 5) aggregation attention -> anchors1
    agg_attn_kernel<<<BATCH * NHEAD, 256, 0, stream>>>(q0_f32, kvagg_bf, seeds, anchors1);

    // 6) self-attention + kv_bcast projection
    self_attn_kernel<<<BATCH, 256, 0, stream>>>(anchors1, w_q_self, w_kv_self, w_kv_bcast, kvb_f32);

    // 7) q_bcast = x @ w_q_bcast  (WMMA, f32 out)
    {
        dim3 grid(ROWS / 64, DIMC / 64);
        wmma_gemm_bf16_kernel<DIMC, DIMC, 0><<<grid, 256, 0, stream>>>(x_bf, wqbT, qb_f32);
    }

    // 8) broadcast attention + depthwise conv + combine -> gl (bf16)
    bcast_combine_kernel<<<(ROWS * NHEAD) / 256, 256, 0, stream>>>(x, qb_f32, kvb_f32, dw_kernel, gl_bf);

    // 9) out = gl @ w_proj  (WMMA, f32 out)
    {
        dim3 grid(ROWS / 64, DIMC / 64);
        wmma_gemm_bf16_kernel<DIMC, DIMC, 0><<<grid, 256, 0, stream>>>(gl_bf, wprojT, out);
    }
}
